// ValueNet_89103391523447
// MI455X (gfx1250) — matrix operand order repaired and checked by reading, not compiled, not run on hardware
//
#include <hip/hip_runtime.h>
#include <hip/hip_bf16.h>
#include <float.h>

#define NN 20000
#define EE 320000
#define BB 32
#define EPN (EE + NN)   // edges + self loops
#define HD 64
#define HH 4
#define ALPHA 0.1f
#define EPSV 1e-5f

typedef __attribute__((ext_vector_type(16))) _Float16 v16h;
typedef __attribute__((ext_vector_type(8)))  _Float16 v8h;
typedef __attribute__((ext_vector_type(8)))  float    v8f;

// ---------- helpers ----------
__device__ __forceinline__ unsigned f2o(float f) {
  unsigned u = __float_as_uint(f);
  return (u & 0x80000000u) ? ~u : (u | 0x80000000u);
}
__device__ __forceinline__ float o2f(unsigned u) {
  u = (u & 0x80000000u) ? (u & 0x7fffffffu) : ~u;
  return __uint_as_float(u);
}
__device__ __forceinline__ float lrelu02(float v) { return v > 0.f ? v : 0.2f * v; }

__global__ void k_fill_f(float* p, float v, int n) {
  int i = blockIdx.x * blockDim.x + threadIdx.x;
  if (i < n) p[i] = v;
}
__global__ void k_fill_u(unsigned* p, unsigned v, int n) {
  int i = blockIdx.x * blockDim.x + threadIdx.x;
  if (i < n) p[i] = v;
}
__global__ void k_copy(float* dst, const float* src, int n) {
  int i = blockIdx.x * blockDim.x + threadIdx.x;
  if (i < n) dst[i] = src[i];
}
__global__ void k_add_inplace(float* a, const float* b, int n) {
  int i = blockIdx.x * blockDim.x + threadIdx.x;
  if (i < n) a[i] += b[i];
}

// ---------- operand packing for WMMA ----------
// A: f32 [Nr,K] -> f16 row-major [Nr,Kp], zero-padded (Kp % 32 == 0)
__global__ void k_packA(const float* __restrict__ A, _Float16* __restrict__ Ah,
                        int Nr, int K, int Kp) {
  long i = (long)blockIdx.x * blockDim.x + threadIdx.x;
  if (i >= (long)Nr * Kp) return;
  int k = (int)(i % Kp);
  long n = i / Kp;
  Ah[i] = (k < K) ? (_Float16)A[n * K + k] : (_Float16)0.f;
}
// B: f32 [K,M] -> f16 swizzled into WMMA B-fragment order:
//   Bp[((kt*M + c)*2 + g)*16 + j] = B[(kt*32 + g*16 + j)*M + c]   (zero if k >= K)
__global__ void k_packB(const float* __restrict__ B, _Float16* __restrict__ Bp,
                        int K, int M, int Kp) {
  int i = blockIdx.x * blockDim.x + threadIdx.x;   // total = Kp*M
  if (i >= Kp * M) return;
  int j = i & 15;
  int g = (i >> 4) & 1;
  int t = i >> 5;
  int c = t % M;
  int kt = t / M;
  int k = kt * 32 + g * 16 + j;
  Bp[i] = (k < K) ? (_Float16)B[(long)k * M + c] : (_Float16)0.f;
}

// ---------- WMMA GEMM: C[Nr x M] = act(Ah[Nr x Kp] @ B + bias) ----------
// Requirements (all satisfied by our shapes): Nr % 16 == 0, M % 64 == 0, Kp % 32 == 0.
// Each wave computes a 16x64 strip with 4 f32 accumulators sharing one A fragment.
// act: 0 = none, 1 = relu, 2 = leaky_relu(0.2)
__global__ void k_gemm_wmma(const _Float16* __restrict__ Ah, const _Float16* __restrict__ Bp,
                            const float* __restrict__ bias, float* __restrict__ C,
                            int Nr, int Kp, int M, int act) {
  const int lane   = threadIdx.x & 31;
  const int waveId = blockIdx.x * (blockDim.x >> 5) + (threadIdx.x >> 5);
  const int strips = M >> 6;
  const int total  = (Nr >> 4) * strips;
  if (waveId >= total) return;                 // wave-uniform exit (EXEC stays all-ones)
  const int tR  = waveId / strips;
  const int tS  = waveId % strips;
  const int l15 = lane & 15;
  const int g   = lane >> 4;                   // lane group 0/1
  const int row = tR * 16 + l15;
  const int c0  = tS * 64 + l15;
  const int Kt  = Kp >> 5;

  const v8h* Ap = (const v8h*)(Ah + (long)row * Kp);
  v8f acc0 = {}, acc1 = {}, acc2 = {}, acc3 = {};
  for (int kt = 0; kt < Kt; ++kt) {
    v8h a0 = Ap[kt * 4 + 2 * g];               // K = kt*32 + g*16 + [0..7]
    v8h a1 = Ap[kt * 4 + 2 * g + 1];           // K = kt*32 + g*16 + 8 + [0..7]
    v16h a = __builtin_shufflevector(a0, a1, 0, 1, 2, 3, 4, 5, 6, 7,
                                     8, 9, 10, 11, 12, 13, 14, 15);
    const _Float16* bb = Bp + (long)kt * M * 32;
    v16h b0 = *(const v16h*)(bb + ((c0 +  0) * 2 + g) * 16);
    v16h b1 = *(const v16h*)(bb + ((c0 + 16) * 2 + g) * 16);
    v16h b2 = *(const v16h*)(bb + ((c0 + 32) * 2 + g) * 16);
    v16h b3 = *(const v16h*)(bb + ((c0 + 48) * 2 + g) * 16);
    acc0 = __builtin_amdgcn_wmma_f32_16x16x32_f16(false, a, false, b0, (short)0, acc0, false, false);
    acc1 = __builtin_amdgcn_wmma_f32_16x16x32_f16(false, a, false, b1, (short)0, acc1, false, false);
    acc2 = __builtin_amdgcn_wmma_f32_16x16x32_f16(false, a, false, b2, (short)0, acc2, false, false);
    acc3 = __builtin_amdgcn_wmma_f32_16x16x32_f16(false, a, false, b3, (short)0, acc3, false, false);
  }
  const int mb = tR * 16 + g * 8;              // first output row of this lane group
#pragma unroll
  for (int s = 0; s < 4; ++s) {
    v8f acc = (s == 0) ? acc0 : (s == 1) ? acc1 : (s == 2) ? acc2 : acc3;
    int col = c0 + s * 16;
    float bv = bias ? bias[col] : 0.f;
#pragma unroll
    for (int r = 0; r < 8; ++r) {
      float v = acc[r] + bv;
      if (act == 1) v = fmaxf(v, 0.f);
      else if (act == 2) v = lrelu02(v);
      C[(long)(mb + r) * M + col] = v;
    }
  }
}

// ---------- input embed: h = [x | J | size_connected] @ W + b ----------
__global__ void k_lin1(const float* __restrict__ x, const float* __restrict__ J,
                       const float* __restrict__ sc, const float* __restrict__ W,
                       const float* __restrict__ b, float* __restrict__ h) {
  int i = blockIdx.x * blockDim.x + threadIdx.x;
  if (i >= NN * HD) return;
  int n = i >> 6, c = i & 63;
  float acc = b[c];
#pragma unroll
  for (int k = 0; k < 8; ++k) acc += x[n * 8 + k] * W[k * HD + c];
  acc += J[n] * W[8 * HD + c] + sc[n] * W[9 * HD + c];
  h[i] = acc;
}

// ---------- GAT attention ----------
__global__ void k_att_scores(const float* __restrict__ xp, const float* __restrict__ aS,
                             const float* __restrict__ aD, float* __restrict__ asrc,
                             float* __restrict__ adst) {
  int i = blockIdx.x * blockDim.x + threadIdx.x;   // NN*HH
  if (i >= NN * HH) return;
  int n = i >> 2, h = i & 3;
  const float* xr = xp + (long)n * 256 + h * 64;
  float s = 0.f, d = 0.f;
#pragma unroll 8
  for (int k = 0; k < 64; ++k) { s += xr[k] * aS[h * 64 + k]; d += xr[k] * aD[h * 64 + k]; }
  asrc[i] = s; adst[i] = d;
}

__device__ __forceinline__ void edge_sd(const int* ei, int e, int& s, int& d) {
  if (e < EE) { s = ei[e]; d = ei[EE + e]; } else { s = e - EE; d = e - EE; }
}

__global__ void k_edge_max(const int* __restrict__ ei, const float* __restrict__ asrc,
                           const float* __restrict__ adst, unsigned* __restrict__ emaxu) {
  int e = blockIdx.x * blockDim.x + threadIdx.x;
  if (e >= EPN) return;
  int s, d; edge_sd(ei, e, s, d);
#pragma unroll
  for (int h = 0; h < HH; ++h) {
    float v = lrelu02(asrc[s * HH + h] + adst[d * HH + h]);
    atomicMax(&emaxu[d * HH + h], f2o(v));
  }
}

__global__ void k_edge_w(const int* __restrict__ ei, const float* __restrict__ asrc,
                         const float* __restrict__ adst, const unsigned* __restrict__ emaxu,
                         float* __restrict__ wbuf, float* __restrict__ denom) {
  int e = blockIdx.x * blockDim.x + threadIdx.x;
  if (e >= EPN) return;
  int s, d; edge_sd(ei, e, s, d);
#pragma unroll
  for (int h = 0; h < HH; ++h) {
    float v = lrelu02(asrc[s * HH + h] + adst[d * HH + h]);
    float w = __expf(v - o2f(emaxu[d * HH + h]));
    wbuf[(long)e * HH + h] = w;
    atomicAdd(&denom[d * HH + h], w);
  }
}

// hsum[dst, d] += sum_h alpha_h * xp[src, h, d]
__global__ void k_edge_aggr(const int* __restrict__ ei, const float* __restrict__ wbuf,
                            const float* __restrict__ denom, const float* __restrict__ xp,
                            float* __restrict__ hsum) {
  long i = (long)blockIdx.x * blockDim.x + threadIdx.x;   // EPN*64
  if (i >= (long)EPN * HD) return;
  int e = (int)(i >> 6), d = (int)(i & 63);
  int s, t; edge_sd(ei, e, s, t);
  float acc = 0.f;
#pragma unroll
  for (int h = 0; h < HH; ++h) {
    float a = wbuf[(long)e * HH + h] / denom[t * HH + h];
    acc += a * xp[(long)s * 256 + h * 64 + d];
  }
  atomicAdd(&hsum[(long)t * HD + d], acc);
}

// ---------- batchnorm ----------
__global__ void k_bn_reduce(const float* __restrict__ t, float* __restrict__ stats) {
  int c = blockIdx.x;                // HD columns
  __shared__ float sh[256], sh2[256];
  float s = 0.f, s2 = 0.f;
  for (int n = threadIdx.x; n < NN; n += blockDim.x) {
    float v = t[(long)n * HD + c];
    s += v; s2 += v * v;
  }
  sh[threadIdx.x] = s; sh2[threadIdx.x] = s2;
  __syncthreads();
  for (int o = blockDim.x >> 1; o > 0; o >>= 1) {
    if (threadIdx.x < o) { sh[threadIdx.x] += sh[threadIdx.x + o]; sh2[threadIdx.x] += sh2[threadIdx.x + o]; }
    __syncthreads();
  }
  if (threadIdx.x == 0) {
    float mu = sh[0] / NN;
    float var = sh2[0] / NN - mu * mu;
    stats[c] = mu;
    stats[HD + c] = rsqrtf(var + EPSV);
  }
}
__global__ void k_bn_apply(float* __restrict__ t, const float* __restrict__ stats,
                           const float* __restrict__ g, const float* __restrict__ b) {
  int i = blockIdx.x * blockDim.x + threadIdx.x;
  if (i >= NN * HD) return;
  int c = i & 63;
  t[i] = (t[i] - stats[c]) * stats[HD + c] * g[c] + b[c];
}

// ---------- APPNP ----------
__global__ void k_deg(const int* __restrict__ ei, float* __restrict__ deg) {
  int e = blockIdx.x * blockDim.x + threadIdx.x;
  if (e >= EPN) return;
  int s, d; edge_sd(ei, e, s, d);
  atomicAdd(&deg[d], 1.f);
}
__global__ void k_appnp_init(float* __restrict__ nxt, const float* __restrict__ x0, int n) {
  int i = blockIdx.x * blockDim.x + threadIdx.x;
  if (i < n) nxt[i] = ALPHA * x0[i];
}
__global__ void k_appnp_edge(const int* __restrict__ ei, const float* __restrict__ deg,
                             const float* __restrict__ cur, float* __restrict__ nxt) {
  long i = (long)blockIdx.x * blockDim.x + threadIdx.x;   // EPN*64
  if (i >= (long)EPN * HD) return;
  int e = (int)(i >> 6), d = (int)(i & 63);
  int s, t; edge_sd(ei, e, s, t);
  float w = (1.f - ALPHA) * rsqrtf(deg[s]) * rsqrtf(deg[t]);
  atomicAdd(&nxt[(long)t * HD + d], w * cur[(long)s * HD + d]);
}

// ---------- pooling ----------
__global__ void k_xg(const float* __restrict__ h, const float* __restrict__ J,
                     const float* __restrict__ sv, const float* __restrict__ inf,
                     float* __restrict__ xg) {
  long i = (long)blockIdx.x * blockDim.x + threadIdx.x;   // NN*67
  if (i >= (long)NN * 67) return;
  int n = (int)(i / 67), c = (int)(i % 67);
  float v;
  if (c < 64) v = h[(long)n * HD + c];
  else if (c == 64) v = J[n];
  else if (c == 65) v = sv[n];
  else v = inf[n];
  xg[i] = v;
}
__global__ void k_rowdot(const float* __restrict__ A, const float* __restrict__ w,
                         const float* __restrict__ b, float* __restrict__ out) {
  int n = blockIdx.x * blockDim.x + threadIdx.x;
  if (n >= NN) return;
  float acc = b[0];
#pragma unroll 8
  for (int k = 0; k < HD; ++k) acc += A[(long)n * HD + k] * w[k];
  out[n] = acc;
}
__global__ void k_seg_max(const float* __restrict__ gate, const int* __restrict__ batch,
                          unsigned* __restrict__ gmaxu) {
  int n = blockIdx.x * blockDim.x + threadIdx.x;
  if (n >= NN) return;
  atomicMax(&gmaxu[batch[n]], f2o(gate[n]));
}
__global__ void k_seg_exp(float* __restrict__ gate, const int* __restrict__ batch,
                          const unsigned* __restrict__ gmaxu, float* __restrict__ esum) {
  int n = blockIdx.x * blockDim.x + threadIdx.x;
  if (n >= NN) return;
  float e = __expf(gate[n] - o2f(gmaxu[batch[n]]));
  gate[n] = e;
  atomicAdd(&esum[batch[n]], e);
}
__global__ void k_pool_acc(const float* __restrict__ gate, const float* __restrict__ esum,
                           const int* __restrict__ batch, const float* __restrict__ m,
                           float* __restrict__ gbuf, int colOff) {
  int i = blockIdx.x * blockDim.x + threadIdx.x;   // NN*64
  if (i >= NN * HD) return;
  int n = i >> 6, d = i & 63;
  float a = gate[n] / esum[batch[n]];
  atomicAdd(&gbuf[batch[n] * 192 + colOff + d], a * m[(long)n * HD + d]);
}

// ---------- final scoring ----------
__global__ void k_xscore(const float* __restrict__ h, const float* __restrict__ gbuf,
                         const int* __restrict__ batch, const float* __restrict__ J,
                         const float* __restrict__ sv, const float* __restrict__ inf,
                         const float* __restrict__ sc, const float* __restrict__ Om,
                         const float* __restrict__ Ph, const float* __restrict__ La,
                         float* __restrict__ xs) {
  long i = (long)blockIdx.x * blockDim.x + threadIdx.x;   // NN*263
  if (i >= (long)NN * 263) return;
  int n = (int)(i / 263), c = (int)(i % 263);
  int b = batch[n];
  float v;
  if (c < 64) v = h[(long)n * HD + c];
  else if (c < 256) v = gbuf[b * 192 + (c - 64)];
  else if (c == 256) v = J[n];
  else if (c == 257) v = sv[n];
  else if (c == 258) v = inf[n];
  else if (c == 259) v = sc[n];
  else if (c == 260) v = Om[b];
  else if (c == 261) v = Ph[b];
  else v = La[b];
  xs[i] = v;
}
__global__ void k_final(const float* __restrict__ s2, const float* __restrict__ w,
                        const float* __restrict__ b, const int* __restrict__ batch,
                        float* __restrict__ out) {
  int n = blockIdx.x * blockDim.x + threadIdx.x;
  if (n >= NN) return;
  float acc = b[0];
#pragma unroll 8
  for (int k = 0; k < HD; ++k) acc += s2[(long)n * HD + k] * w[k];
  float s = 1.f / (1.f + __expf(-acc));
  atomicAdd(&out[batch[n]], s);
}

// ---------- host ----------
static inline long cdiv(long a, long b) { return (a + b - 1) / b; }

extern "C" void kernel_launch(void* const* d_in, const int* in_sizes, int n_in,
                              void* d_out, int out_size, void* d_ws, size_t ws_size,
                              hipStream_t stream) {
  // input mapping (setup_inputs insertion order; params dict flattened in insertion order)
  const float* x    = (const float*)d_in[0];
  const int*   ei   = (const int*)d_in[1];
  const int*   batch= (const int*)d_in[2];
  const float* J    = (const float*)d_in[3];
  const float* sv   = (const float*)d_in[4];
  const float* inf  = (const float*)d_in[5];
  const float* sc   = (const float*)d_in[6];
  const float* Om   = (const float*)d_in[7];
  const float* Ph   = (const float*)d_in[8];
  const float* La   = (const float*)d_in[9];
  const float* lin1_W = (const float*)d_in[10];
  const float* lin1_b = (const float*)d_in[11];
  const float* fin_lin3_W = (const float*)d_in[72];
  const float* fin_lin3_b = (const float*)d_in[73];
  const float* fin_bn1_g  = (const float*)d_in[74];
  const float* fin_bn1_b  = (const float*)d_in[75];
  const float* fin_lin4_W = (const float*)d_in[76];
  const float* fin_lin4_b = (const float*)d_in[77];
  const float* fin_bn2_g  = (const float*)d_in[78];
  const float* fin_bn2_b  = (const float*)d_in[79];
  const float* fin_lin5_W = (const float*)d_in[80];
  const float* fin_lin5_b = (const float*)d_in[81];

  float* out = (float*)d_out;

  // ---- workspace carve-out (f32 part) ----
  float* ws = (float*)d_ws;
  size_t off = 0;
  auto alloc = [&](size_t n) { float* p = ws + off; off += n; return p; };
  float* hbuf  = alloc((size_t)NN * HD);
  float* t1    = alloc((size_t)NN * HD);
  float* t2    = alloc((size_t)NN * HD);
  float* hsum  = alloc((size_t)NN * HD);        // also x0 during APPNP
  float* big   = alloc((size_t)NN * 263);       // xp (N,256) then x_score (N,263)
  float* wbuf  = alloc((size_t)EPN * HH);       // edge weights; reused as m (N,64)
  float* xg    = alloc((size_t)NN * 67);
  float* asrc  = alloc((size_t)NN * HH);
  float* adst  = alloc((size_t)NN * HH);
  unsigned* emaxu = (unsigned*)alloc((size_t)NN * HH);
  float* denom = alloc((size_t)NN * HH);
  float* deg   = alloc((size_t)NN);
  float* gate  = alloc((size_t)NN);
  float* stats = alloc(2 * HD);
  float* esum  = alloc(BB);
  unsigned* gmaxu = (unsigned*)alloc(BB);
  float* gbuf  = alloc((size_t)BB * 192);
  // ---- f16 pool (32B-aligned) ----
  off = (off + 15) & ~(size_t)15;               // 64-byte alignment in bytes
  _Float16* hpool = (_Float16*)(ws + off);
  size_t hoff = 0;
  auto allocH = [&](size_t n) { _Float16* p = hpool + hoff; hoff += (n + 31) & ~(size_t)31; return p; };
  _Float16* Ah  = allocH((size_t)NN * 288);     // activations up to Kp=288
  _Float16* Ah2 = allocH((size_t)NN * 64);      // 64-wide activations
  _Float16* BpGat[3], *BpAl1[3], *BpAl2[3], *BpAl3[3];
  _Float16* BpGW1[3], *BpNW1[3], *BpNW2[3];
  for (int L = 0; L < 3; ++L) {
    BpGat[L] = allocH(64 * 256);
    BpAl1[L] = allocH(64 * 64);
    BpAl2[L] = allocH(64 * 64);
    BpAl3[L] = allocH(64 * 64);
    BpGW1[L] = allocH(96 * 64);
    BpNW1[L] = allocH(96 * 64);
    BpNW2[L] = allocH(64 * 64);
  }
  _Float16* BpL3 = allocH(288 * 64);
  _Float16* BpL4 = allocH(64 * 64);
  (void)ws_size; (void)n_in; (void)in_sizes; (void)out_size;

  const int T = 256;
  const unsigned NEG_MAX = 0x00800000u;   // f2o(-FLT_MAX)
  const long nhd = (long)NN * HD;
  const long ned = (long)EPN * HD;

  auto packB = [&](const float* B, _Float16* Bp, int K, int M, int Kp) {
    k_packB<<<dim3(cdiv((long)Kp * M, T)), dim3(T), 0, stream>>>(B, Bp, K, M, Kp);
  };
  auto packA = [&](const float* A, _Float16* Adst, int K, int Kp) {
    k_packA<<<dim3(cdiv((long)NN * Kp, T)), dim3(T), 0, stream>>>(A, Adst, NN, K, Kp);
  };
  auto gemm = [&](const _Float16* Ap, int Kp, const _Float16* Bp, const float* bias,
                  float* C, int M, int act) {
    long waves = (NN / 16) * (M / 64);
    k_gemm_wmma<<<dim3(cdiv(waves, 8)), dim3(256), 0, stream>>>(Ap, Bp, bias, C, NN, Kp, M, act);
  };
  auto bn = [&](float* buf, const float* g, const float* b) {
    k_bn_reduce<<<dim3(HD), dim3(256), 0, stream>>>(buf, stats);
    k_bn_apply<<<dim3(cdiv(nhd, T)), dim3(T), 0, stream>>>(buf, stats, g, b);
  };

  // ---- pre-pack all weight matrices into WMMA B-fragment order ----
  for (int L = 0; L < 3; ++L) {
    int pb = 12 + L * 20;
    packB((const float*)d_in[pb + 0], BpGat[L], 64, 256, 64);
    packB((const float*)d_in[pb + 3], BpAl1[L], 64, 64, 64);
    packB((const float*)d_in[pb + 6], BpAl2[L], 64, 64, 64);
    packB((const float*)d_in[pb + 8], BpAl3[L], 64, 64, 64);
    packB((const float*)d_in[pb + 12], BpGW1[L], 67, 64, 96);
    packB((const float*)d_in[pb + 16], BpNW1[L], 67, 64, 96);
    packB((const float*)d_in[pb + 18], BpNW2[L], 64, 64, 64);
  }
  packB(fin_lin3_W, BpL3, 263, 64, 288);
  packB(fin_lin4_W, BpL4, 64, 64, 64);

  // ---- input embedding ----
  k_lin1<<<dim3(cdiv(nhd, T)), dim3(T), 0, stream>>>(x, J, sc, lin1_W, lin1_b, hbuf);

  // ---- 3 attention layers ----
  for (int L = 0; L < 3; ++L) {
    int pb = 12 + L * 20;
    const float* a_src = (const float*)d_in[pb + 1];
    const float* a_dst = (const float*)d_in[pb + 2];
    const float* bn1_g = (const float*)d_in[pb + 4];
    const float* bn1_b = (const float*)d_in[pb + 5];
    const float* al2_b = (const float*)d_in[pb + 7];
    const float* al3_b = (const float*)d_in[pb + 9];
    const float* bn2_g = (const float*)d_in[pb + 10];
    const float* bn2_b = (const float*)d_in[pb + 11];

    packA(hbuf, Ah2, 64, 64);
    gemm(Ah2, 64, BpGat[L], nullptr, big, 256, 0);                         // xp
    k_att_scores<<<dim3(cdiv((long)NN * HH, T)), dim3(T), 0, stream>>>(big, a_src, a_dst, asrc, adst);
    k_fill_u<<<dim3(cdiv((long)NN * HH, T)), dim3(T), 0, stream>>>(emaxu, NEG_MAX, NN * HH);
    k_fill_f<<<dim3(cdiv((long)NN * HH, T)), dim3(T), 0, stream>>>(denom, 0.f, NN * HH);
    k_fill_f<<<dim3(cdiv(nhd, T)), dim3(T), 0, stream>>>(hsum, 0.f, (int)nhd);
    k_edge_max<<<dim3(cdiv(EPN, T)), dim3(T), 0, stream>>>(ei, asrc, adst, emaxu);
    k_edge_w<<<dim3(cdiv(EPN, T)), dim3(T), 0, stream>>>(ei, asrc, adst, emaxu, wbuf, denom);
    k_edge_aggr<<<dim3(cdiv(ned, T)), dim3(T), 0, stream>>>(ei, wbuf, denom, big, hsum);

    packA(hsum, Ah2, 64, 64);
    gemm(Ah2, 64, BpAl1[L], nullptr, t1, 64, 0);                           // lin1
    k_add_inplace<<<dim3(cdiv(nhd, T)), dim3(T), 0, stream>>>(t1, hbuf, (int)nhd);
    bn(t1, bn1_g, bn1_b);
    packA(t1, Ah2, 64, 64);
    gemm(Ah2, 64, BpAl2[L], al2_b, t2, 64, 1);                             // relu MLP
    packA(t2, Ah2, 64, 64);
    gemm(Ah2, 64, BpAl3[L], al3_b, hbuf, 64, 0);
    k_add_inplace<<<dim3(cdiv(nhd, T)), dim3(T), 0, stream>>>(hbuf, t1, (int)nhd);
    bn(hbuf, bn2_g, bn2_b);                                                // layer out in hbuf
  }

  // ---- APPNP ----
  k_fill_f<<<dim3(cdiv(NN, T)), dim3(T), 0, stream>>>(deg, 0.f, NN);
  k_deg<<<dim3(cdiv(EPN, T)), dim3(T), 0, stream>>>(ei, deg);
  k_copy<<<dim3(cdiv(nhd, T)), dim3(T), 0, stream>>>(hsum, hbuf, (int)nhd); // x0
  float* cur = hbuf; float* nxt = t1;
  for (int it = 0; it < 10; ++it) {
    k_appnp_init<<<dim3(cdiv(nhd, T)), dim3(T), 0, stream>>>(nxt, hsum, (int)nhd);
    k_appnp_edge<<<dim3(cdiv(ned, T)), dim3(T), 0, stream>>>(ei, deg, cur, nxt);
    float* tmp = cur; cur = nxt; nxt = tmp;
  }
  // 10 iterations (even) -> cur == hbuf

  // ---- pooling ----
  k_xg<<<dim3(cdiv((long)NN * 67, T)), dim3(T), 0, stream>>>(cur, J, sv, inf, xg);
  k_fill_f<<<dim3(1), dim3(T), 0, stream>>>(gbuf, 0.f, BB * 192);
  packA(xg, Ah, 67, 96);                         // xg fragment reused by all 3 pools
  float* mbuf = wbuf;                            // reuse edge-weight buffer (EPN*4 >= NN*64)
  for (int P = 0; P < 3; ++P) {
    int pb = 12 + P * 20 + 12;
    const float* gb1 = (const float*)d_in[pb + 1];
    const float* gW2 = (const float*)d_in[pb + 2];
    const float* gb2 = (const float*)d_in[pb + 3];
    const float* nb1 = (const float*)d_in[pb + 5];
    const float* nb2 = (const float*)d_in[pb + 7];

    gemm(Ah, 96, BpGW1[P], gb1, t2, 64, 1);
    k_rowdot<<<dim3(cdiv(NN, T)), dim3(T), 0, stream>>>(t2, gW2, gb2, gate);
    k_fill_u<<<dim3(1), dim3(32), 0, stream>>>(gmaxu, NEG_MAX, BB);
    k_fill_f<<<dim3(1), dim3(32), 0, stream>>>(esum, 0.f, BB);
    k_seg_max<<<dim3(cdiv(NN, T)), dim3(T), 0, stream>>>(gate, batch, gmaxu);
    k_seg_exp<<<dim3(cdiv(NN, T)), dim3(T), 0, stream>>>(gate, batch, gmaxu, esum);
    gemm(Ah, 96, BpNW1[P], nb1, t2, 64, 1);
    packA(t2, Ah2, 64, 64);
    gemm(Ah2, 64, BpNW2[P], nb2, mbuf, 64, 0);
    k_pool_acc<<<dim3(cdiv(nhd, T)), dim3(T), 0, stream>>>(gate, esum, batch, mbuf, gbuf, P * 64);
  }

  // ---- final scoring head ----
  k_xscore<<<dim3(cdiv((long)NN * 263, T)), dim3(T), 0, stream>>>(
      cur, gbuf, batch, J, sv, inf, sc, Om, Ph, La, big);
  packA(big, Ah, 263, 288);
  gemm(Ah, 288, BpL3, fin_lin3_b, t1, 64, 2);    // leaky 0.2
  bn(t1, fin_bn1_g, fin_bn1_b);
  packA(t1, Ah2, 64, 64);
  gemm(Ah2, 64, BpL4, fin_lin4_b, t2, 64, 2);
  bn(t2, fin_bn2_g, fin_bn2_b);
  k_fill_f<<<dim3(1), dim3(32), 0, stream>>>(out, 0.f, BB);
  k_final<<<dim3(cdiv(NN, T)), dim3(T), 0, stream>>>(t2, fin_lin5_W, fin_lin5_b, batch, out);
}
